// BiRNN_39719857553991
// MI455X (gfx1250) — matrix operand order repaired and checked by reading, not compiled, not run on hardware
//
#include <hip/hip_runtime.h>
#include <cstdint>
#include <cstddef>

// ---------------------------------------------------------------------------
// Bidirectional LSTM (TF BasicLSTMCell), B=64, T=128, F=H=512, forget_bias=1.
// Phase A: gates_x = x @ W_x + b  (bf16 WMMA, f32 accum), fully parallel.
// Phase B: per-direction sequential scan, one 32-wave workgroup per direction,
//          h @ W_h via v_wmma_f32_16x16x32_bf16, c-state kept in VGPRs.
// ---------------------------------------------------------------------------

typedef __bf16 bf16;
typedef __attribute__((ext_vector_type(8)))  __bf16 v8bf;
typedef __attribute__((ext_vector_type(16))) __bf16 v16bf;
typedef __attribute__((ext_vector_type(8)))  float  v8f;

#define BB   64      // batch
#define TT   128     // time steps
#define FF   512     // input features
#define HH   512     // hidden
#define NG   2048    // 4*H gate width
#define WROW 1024    // F+H, row length of transposed weights

static __device__ __forceinline__ v16bf cat16(v8bf lo, v8bf hi) {
  return __builtin_shufflevector(lo, hi, 0,1,2,3,4,5,6,7,8,9,10,11,12,13,14,15);
}
static __device__ __forceinline__ float sigf(float x) {
  return 1.0f / (1.0f + __expf(-x));
}

// --------------------------- f32 -> bf16 convert ---------------------------
__global__ void cvt_x_kernel(const float* __restrict__ x, bf16* __restrict__ xb, int n) {
  int i = blockIdx.x * 256 + threadIdx.x;
  if (i < n) xb[i] = (bf16)x[i];
}

// Transpose + convert W[dir][k][n] (k=0..1023, n=0..2047) -> Wt[dir][n][k] bf16.
// Rows of Wt are contiguous along K: first 512 = W_x part, last 512 = W_h part.
__global__ void cvt_w_kernel(const float* __restrict__ Wfw, const float* __restrict__ Wbw,
                             bf16* __restrict__ wt) {
  int i = blockIdx.x * 256 + threadIdx.x;   // 0 .. 2*2048*1024-1 (2^22)
  int dir = i >> 21;
  int r   = i & ((1 << 21) - 1);
  int n   = r >> 10;
  int k   = r & 1023;
  const float* W = dir ? Wbw : Wfw;
  wt[i] = (bf16)W[(size_t)k * NG + n];
}

// ------------------------ Phase A: gates_x = x @ W_x + b -------------------
// Per wave: 64(M) x 32(N) output tile, K=512 in steps of 32.
__global__ __launch_bounds__(256) void
gemm_x_kernel(const bf16* __restrict__ xb, const bf16* __restrict__ wt,
              const float* __restrict__ bias_fw, const float* __restrict__ bias_bw,
              float* __restrict__ gates) {
  const int dir  = blockIdx.y;
  const int w    = threadIdx.x >> 5;
  const int lane = threadIdx.x & 31;
  const int half = lane >> 4;
  const int l16  = lane & 15;
  const int gid  = blockIdx.x * 8 + w;          // 0..8191 wave tiles
  const int mT   = (gid >> 6) * 64;             // M base (M total 8192)
  const int nT   = (gid & 63) * 32;             // N base (N total 2048)

  const bf16*  wd = wt    + (size_t)dir * ((size_t)NG * WROW);
  const float* bs = dir ? bias_bw : bias_fw;
  float*       gd = gates + (size_t)dir * ((size_t)TT * BB * NG);

  v8f acc[4][2];
  #pragma unroll
  for (int mt = 0; mt < 4; ++mt)
    #pragma unroll
    for (int nt = 0; nt < 2; ++nt)
      #pragma unroll
      for (int v = 0; v < 8; ++v) acc[mt][nt][v] = 0.0f;

  // A rows: gates row m = t*64 + b  ->  x row = b*T + t
  const bf16* arow[4];
  #pragma unroll
  for (int mt = 0; mt < 4; ++mt) {
    int m = mT + mt * 16 + l16;
    int t = m >> 6, b = m & 63;
    arow[mt] = xb + ((size_t)(b * TT + t)) * FF;
  }
  const bf16* brow[2];
  #pragma unroll
  for (int nt = 0; nt < 2; ++nt)
    brow[nt] = wd + (size_t)(nT + nt * 16 + l16) * WROW;   // W_x part: k in [0,512)

  for (int k0 = 0; k0 < FF; k0 += 32) {
    v16bf a[4], bb[2];
    #pragma unroll
    for (int mt = 0; mt < 4; ++mt) {
      v8bf lo = *(const v8bf*)(arow[mt] + k0 + half * 8);        // K = 8*half+0..7
      v8bf hi = *(const v8bf*)(arow[mt] + k0 + 16 + half * 8);   // K = 16+8*half+0..7
      a[mt] = cat16(lo, hi);
    }
    #pragma unroll
    for (int nt = 0; nt < 2; ++nt) {
      v8bf lo = *(const v8bf*)(brow[nt] + k0 + half * 8);        // K = 8*half+0..7
      v8bf hi = *(const v8bf*)(brow[nt] + k0 + 16 + half * 8);   // K = 16+8*half+0..7
      bb[nt] = cat16(lo, hi);
    }
    #pragma unroll
    for (int mt = 0; mt < 4; ++mt)
      #pragma unroll
      for (int nt = 0; nt < 2; ++nt)
        acc[mt][nt] = __builtin_amdgcn_wmma_f32_16x16x32_bf16(
            false, a[mt], false, bb[nt], (short)0, acc[mt][nt], false, false);
  }

  #pragma unroll
  for (int nt = 0; nt < 2; ++nt) {
    int n = nT + nt * 16 + l16;
    float bv = bs[n];
    #pragma unroll
    for (int mt = 0; mt < 4; ++mt)
      #pragma unroll
      for (int v = 0; v < 8; ++v) {
        int m = mT + mt * 16 + half * 8 + v;
        gd[(size_t)m * NG + n] = acc[mt][nt][v] + bv;
      }
  }
}

// ---------------------- Phase B: sequential LSTM scan ----------------------
// One block (32 wave32) per direction. Wave w owns h-columns [16w,16w+16) and
// computes all 4 gate columns for those -> gate math + c-state stay in VGPRs.
// h_prev staged in LDS (64x512 bf16 = 64KB) with XOR chunk swizzle.
__global__ __launch_bounds__(1024) void
lstm_scan_kernel(const float* __restrict__ gates, const bf16* __restrict__ wt,
                 float* __restrict__ out) {
  __shared__ __align__(16) bf16 h_lds[BB * HH];   // 64KB

  const int dir  = blockIdx.x;
  const int tid  = threadIdx.x;
  const int w    = tid >> 5;
  const int lane = tid & 31;
  const int half = lane >> 4;
  const int l16  = lane & 15;
  const int ncol = w * 16 + l16;                  // h column this lane owns (0..511)

  const float* gd = gates + (size_t)dir * ((size_t)TT * BB * NG);
  const bf16*  wd = wt    + (size_t)dir * ((size_t)NG * WROW);

  // B rows for the 4 gates, W_h part (k offset +512 within each row)
  const bf16* brow[4];
  #pragma unroll
  for (int g = 0; g < 4; ++g)
    brow[g] = wd + (size_t)(g * HH + ncol) * WROW + FF;

  for (int i = tid; i < BB * HH; i += 1024) h_lds[i] = (bf16)0.0f;

  v8f cst[4];
  #pragma unroll
  for (int mt = 0; mt < 4; ++mt)
    #pragma unroll
    for (int v = 0; v < 8; ++v) cst[mt][v] = 0.0f;

  __syncthreads();

  for (int s = 0; s < TT; ++s) {
    const int t = dir ? (TT - 1 - s) : s;
    const float* gt = gd + (size_t)t * BB * NG;

    // Init accumulators from precomputed x-gates (acts as the C matrix).
    v8f acc[4][4];
    #pragma unroll
    for (int g = 0; g < 4; ++g)
      #pragma unroll
      for (int mt = 0; mt < 4; ++mt)
        #pragma unroll
        for (int v = 0; v < 8; ++v) {
          int m = mt * 16 + half * 8 + v;
          acc[g][mt][v] = gt[(size_t)m * NG + g * HH + ncol];
        }

    // h_prev @ W_h
    for (int k0 = 0; k0 < HH; k0 += 32) {
      const int c0 = k0 >> 3;                     // chunk index base (8 elems/chunk)
      v16bf a[4];
      #pragma unroll
      for (int mt = 0; mt < 4; ++mt) {
        int m   = mt * 16 + l16;
        int msw = m & 7;                          // XOR bank swizzle per row
        const bf16* rb = &h_lds[m * HH];
        v8bf lo = *(const v8bf*)(rb + (((c0 + half)     ^ msw) << 3));
        v8bf hi = *(const v8bf*)(rb + (((c0 + 2 + half) ^ msw) << 3));
        a[mt] = cat16(lo, hi);
      }
      #pragma unroll
      for (int g = 0; g < 4; ++g) {
        v8bf lo = *(const v8bf*)(brow[g] + k0 + half * 8);
        v8bf hi = *(const v8bf*)(brow[g] + k0 + 16 + half * 8);
        v16bf bbf = cat16(lo, hi);
        #pragma unroll
        for (int mt = 0; mt < 4; ++mt)
          acc[g][mt] = __builtin_amdgcn_wmma_f32_16x16x32_bf16(
              false, a[mt], false, bbf, (short)0, acc[g][mt], false, false);
      }
    }

    __syncthreads();   // everyone done reading h_lds for this step

    // Gate nonlinearity + state update, all in registers; write h to out + LDS.
    #pragma unroll
    for (int mt = 0; mt < 4; ++mt)
      #pragma unroll
      for (int v = 0; v < 8; ++v) {
        float gi = acc[0][mt][v];
        float gj = acc[1][mt][v];
        float gf = acc[2][mt][v];
        float go = acc[3][mt][v];
        float c2 = cst[mt][v] * sigf(gf + 1.0f) + sigf(gi) * tanhf(gj);
        cst[mt][v] = c2;
        float h = tanhf(c2) * sigf(go);
        int b = mt * 16 + half * 8 + v;           // batch row
        out[(size_t)b * (TT * 2 * HH) + (size_t)t * (2 * HH) + dir * HH + ncol] = h;
        int chunk = (ncol >> 3) ^ (b & 7);
        h_lds[b * HH + (chunk << 3) + (ncol & 7)] = (bf16)h;
      }

    __syncthreads();   // h_lds fully updated before next step's reads
  }
}

// --------------------------------- launch ----------------------------------
extern "C" void kernel_launch(void* const* d_in, const int* in_sizes, int n_in,
                              void* d_out, int out_size, void* d_ws, size_t ws_size,
                              hipStream_t stream) {
  (void)in_sizes; (void)n_in; (void)out_size; (void)ws_size;
  const float* x   = (const float*)d_in[0];
  const float* Wfw = (const float*)d_in[1];
  const float* bfw = (const float*)d_in[2];
  const float* Wbw = (const float*)d_in[3];
  const float* bbw = (const float*)d_in[4];
  float* out = (float*)d_out;

  char* ws = (char*)d_ws;
  bf16*  xb    = (bf16*)ws;                            // 8 MB  : x bf16 [B][T][F]
  bf16*  wt    = (bf16*)(ws + (size_t)(8u << 20));     // 8 MB  : Wt bf16 [2][2048][1024]
  float* gates = (float*)(ws + (size_t)(16u << 20));   // 128 MB: [2][T*B][2048] f32

  const int nx = BB * TT * FF;                         // 4,194,304
  cvt_x_kernel<<<nx / 256, 256, 0, stream>>>(x, xb, nx);
  cvt_w_kernel<<<(2 * NG * WROW) / 256, 256, 0, stream>>>(Wfw, Wbw, wt);
  gemm_x_kernel<<<dim3(1024, 2), 256, 0, stream>>>(xb, wt, bfw, bbw, gates);
  lstm_scan_kernel<<<2, 1024, 0, stream>>>(gates, wt, out);
}
